// LinearAttention_1511828488338
// MI455X (gfx1250) — matrix operand order repaired and checked by reading, not compiled, not run on hardware
//
#include <hip/hip_runtime.h>

// -------- CDNA5 (gfx1250, wave32) WMMA types --------
typedef __attribute__((ext_vector_type(16))) _Float16 v16h;
typedef __attribute__((ext_vector_type(8)))  _Float16 v8h;
typedef __attribute__((ext_vector_type(4)))  _Float16 v4h;
typedef __attribute__((ext_vector_type(8)))  float    v8f;
typedef int v4i_vs __attribute__((vector_size(16)));   // matches async builtin param

// -------- Problem constants (reference: B,S,H,NH,HD = 4,4096,1024,16,64) ----
constexpr int Bc  = 4;
constexpr int Sc  = 4096;
constexpr int Hc  = 1024;
constexpr int NHc = 16;
constexpr int HDc = 64;
constexpr int MTOK = Bc * Sc;        // 16384 token rows
constexpr int KSLICE = 16;           // K-split of the kv reduction

// -------- optional CDNA5 async global->LDS path (ASYNCcnt) ------------------
#if defined(__has_builtin)
#  if __has_builtin(__builtin_amdgcn_global_load_async_to_lds_b128) && \
      __has_builtin(__builtin_amdgcn_s_wait_asynccnt)
#    define USE_ASYNC_LDS 1
#  endif
#endif
#ifndef USE_ASYNC_LDS
#  define USE_ASYNC_LDS 0
#endif

__device__ __forceinline__ void copy16B_lds(const _Float16* g, _Float16* l) {
#if USE_ASYNC_LDS
  // emits global_load_async_to_lds_b128 (tracked with ASYNCcnt)
  __builtin_amdgcn_global_load_async_to_lds_b128(
      (__attribute__((address_space(1))) v4i_vs*)g,
      (__attribute__((address_space(3))) v4i_vs*)l, 0, 0);
#else
  *(v8h*)l = *(const v8h*)g;
#endif
}
__device__ __forceinline__ void async_join() {
#if USE_ASYNC_LDS
  __builtin_amdgcn_s_wait_asynccnt(0);   // s_wait_asynccnt 0
#endif
}

// -------- WMMA fragment helpers ---------------------------------------------
__device__ __forceinline__ v16h make_frag(v8h lo, v8h hi) {
  v16h r;
#pragma unroll
  for (int i = 0; i < 8; ++i) { r[i] = lo[i]; r[i + 8] = hi[i]; }
  return r;
}

// A fragment (16x32 f16, M=row, K contiguous). rowk -> &tile[row][k0], kg = lane>>4.
__device__ __forceinline__ v16h frag_a(const _Float16* rowk, int kg) {
  v8h lo = *(const v8h*)(rowk + kg * 8);
  v8h hi = *(const v8h*)(rowk + 16 + kg * 8);
  return make_frag(lo, hi);
}
// B fragment (32x16 f16, stored K-major per column). colk -> &tileT[col][k0].
__device__ __forceinline__ v16h frag_b(const _Float16* colk, int kg) {
  v8h lo = *(const v8h*)(colk + kg * 8);
  v8h hi = *(const v8h*)(colk + 16 + kg * 8);
  return make_frag(lo, hi);
}

__device__ __forceinline__ v8f wmma_f16(v16h a, v16h b, v8f c) {
  return __builtin_amdgcn_wmma_f32_16x16x32_f16(false, a, false, b, (short)0, c,
                                                false, false);
}

__device__ __forceinline__ float elu1(float x) {
  return x > 0.f ? x + 1.f : __expf(x);   // elu(x)+1
}

__device__ __forceinline__ v4h load4cvt(const float* p) {
  const float4 t = *(const float4*)p;
  v4h r;
  r[0] = (_Float16)t.x; r[1] = (_Float16)t.y;
  r[2] = (_Float16)t.z; r[3] = (_Float16)t.w;
  return r;
}

// -------- tile stagers (128 x 32 tile, LDS stride 40) ------------------------
constexpr int LDT_G = 40;

// f32 source: convert in registers while staging.
__device__ __forceinline__ void stage_tile(const float* __restrict__ src, long ld,
                                           _Float16* dst, int tid) {
  const int r0 = tid >> 3, c = (tid & 7) * 4;
#pragma unroll
  for (int i = 0; i < 4; ++i) {
    const int r = r0 + i * 32;
    *(v4h*)&dst[r * LDT_G + c] = load4cvt(src + (long)r * ld + c);
  }
}
// f16 source: straight 16B global->LDS copies (async path when available).
__device__ __forceinline__ void stage_tile(const _Float16* __restrict__ src, long ld,
                                           _Float16* dst, int tid) {
#pragma unroll
  for (int i = 0; i < 2; ++i) {
    const int id = tid + i * 256;
    const int r = id >> 2, c = (id & 3) * 8;
    copy16B_lds(src + (long)r * ld + c, &dst[r * LDT_G + c]);
  }
}

// ============================================================================
// Kernel 0: one-time f32 -> f16 weight conversion (removes per-block cvt work)
// ============================================================================
__global__ __launch_bounds__(256) void cvt_w_kernel(const float* __restrict__ src,
                                                    _Float16* __restrict__ dst) {
  const long i = (long)blockIdx.x * 256 + threadIdx.x;   // one float4 per thread
  *(v4h*)(dst + i * 4) = load4cvt(src + i * 4);
}

// ============================================================================
// Kernel 1/4: Y[M,N] = act(X[M,K] . W16[N,K]^T + bias[N])
// Block = 256 thr (8 waves), tile 128x128, K-step 32; wave owns 32x64 (8 WMMA).
// ============================================================================
template <typename TIN, typename TOUT, bool ACT>
__global__ __launch_bounds__(256) void gemm_xwt(
    const TIN* __restrict__ X, const _Float16* __restrict__ W,
    const float* __restrict__ bias, TOUT* __restrict__ Y,
    int M, int N, int K)
{
  constexpr int BM = 128, BN = 128, BK = 32, LDT = LDT_G;
  __shared__ __align__(16) _Float16 Xs[BM * LDT];
  __shared__ __align__(16) _Float16 Ws[BN * LDT];

  const int tid  = threadIdx.x;
  const int lane = tid & 31;
  const int wave = __builtin_amdgcn_readfirstlane(tid >> 5);  // scalar wave id
  const int wr   = wave & 3;      // 32-row slice
  const int wc   = wave >> 2;     // 64-col slice
  const int kg   = lane >> 4;     // WMMA K-group / C row-half
  const int lrow = lane & 15;

  const long m0 = (long)blockIdx.y * BM;
  const long n0 = (long)blockIdx.x * BN;

  v8f acc[2][4] = {};

  for (int k0 = 0; k0 < K; k0 += BK) {
    if (k0 + BK < K) {  // next-tile hints -> global_prefetch_b8
      __builtin_prefetch(X + (m0 + (tid >> 3)) * (long)K + k0 + BK, 0, 1);
      __builtin_prefetch(W + (n0 + (tid >> 3)) * (long)K + k0 + BK, 0, 1);
    }
    __syncthreads();
    stage_tile(X + m0 * K + k0, K, Xs, tid);
    stage_tile(W + n0 * K + k0, K, Ws, tid);
    async_join();
    __syncthreads();

    const v16h a0 = frag_a(&Xs[(wr * 32 + lrow) * LDT], kg);
    const v16h a1 = frag_a(&Xs[(wr * 32 + 16 + lrow) * LDT], kg);
#pragma unroll
    for (int ct = 0; ct < 4; ++ct) {
      const v16h bf = frag_b(&Ws[(wc * 64 + ct * 16 + lrow) * LDT], kg);
      acc[0][ct] = wmma_f16(a0, bf, acc[0][ct]);
      acc[1][ct] = wmma_f16(a1, bf, acc[1][ct]);
    }
  }

  // Epilogue. C/D layout: VGPR r -> M = r + 8*(lane>>4), N = lane&15.
#pragma unroll
  for (int rt = 0; rt < 2; ++rt) {
#pragma unroll
    for (int ct = 0; ct < 4; ++ct) {
      const long col = n0 + wc * 64 + ct * 16 + lrow;
      const float bvv = bias[col];
#pragma unroll
      for (int r = 0; r < 8; ++r) {
        const long row = m0 + wr * 32 + rt * 16 + kg * 8 + r;
        float val = acc[rt][ct][r] + bvv;
        if (ACT) val = elu1(val);
        Y[row * (long)N + col] = (TOUT)val;
      }
    }
  }
}

// ============================================================================
// Kernel 2a: K-sliced kv partial: part[slice][bh][64][64] over 256 t's/slice.
// Block = (bh, slice); 8 waves = 4 d-slices x 2 chunk-parities; LDS pair-reduce.
// ============================================================================
__global__ __launch_bounds__(256) void kv_reduce_kernel(
    const _Float16* __restrict__ kf, const _Float16* __restrict__ vv,
    float* __restrict__ part)
{
  constexpr int LDT = LDT_G;
  __shared__ __align__(16) _Float16 kts[HDc * LDT];   // [d][t]
  __shared__ __align__(16) _Float16 vts[HDc * LDT];   // [e][t]
  __shared__ float red[HDc * HDc];

  const int bh = blockIdx.x;
  const int slice = blockIdx.y;
  const int b = bh / NHc, h = bh % NHc;
  const long headoff = (long)b * Sc * Hc + (long)h * HDc;

  const int tid = threadIdx.x, lane = tid & 31;
  const int wave = __builtin_amdgcn_readfirstlane(tid >> 5);
  const int ms   = wave & 3;     // d slice of 16
  const int kpar = wave >> 2;    // chunk parity this wave accumulates (scalar!)
  const int kg = lane >> 4, lrow = lane & 15;

  const int lt  = tid >> 3;        // t within 32-chunk
  const int ld8 = (tid & 7) * 8;   // feature offset

  constexpr int CHUNKS = (Sc / 32) / KSLICE;   // 8 chunks of 32 t's
  v8f acc[4] = {};

  for (int cc = 0; cc < CHUNKS; ++cc) {
    const int c = slice * CHUNKS + cc;
    const long g = headoff + (long)(c * 32 + lt) * Hc + ld8;
    const v8h kk = *(const v8h*)(kf + g);
    const v8h vk = *(const v8h*)(vv + g);
    __syncthreads();
#pragma unroll
    for (int i = 0; i < 8; ++i) {       // transpose on store: [t][d] -> [d][t]
      kts[(ld8 + i) * LDT + lt] = kk[i];
      vts[(ld8 + i) * LDT + lt] = vk[i];
    }
    __syncthreads();
    if ((cc & 1) == kpar) {             // scalar branch: EXEC stays all-ones
      const v16h a = frag_a(&kts[(ms * 16 + lrow) * LDT], kg);
#pragma unroll
      for (int ct = 0; ct < 4; ++ct) {
        const v16h bf = frag_b(&vts[(ct * 16 + lrow) * LDT], kg);
        acc[ct] = wmma_f16(a, bf, acc[ct]);
      }
    }
  }

  if (kpar == 1) {
#pragma unroll
    for (int ct = 0; ct < 4; ++ct)
#pragma unroll
      for (int r = 0; r < 8; ++r)
        red[(ms * 16 + kg * 8 + r) * HDc + ct * 16 + lrow] = acc[ct][r];
  }
  __syncthreads();
  if (kpar == 0) {
    float* dst = part + ((long)slice * (Bc * NHc) + bh) * (HDc * HDc);
#pragma unroll
    for (int ct = 0; ct < 4; ++ct)
#pragma unroll
      for (int r = 0; r < 8; ++r) {
        const int d = ms * 16 + kg * 8 + r;
        const int e = ct * 16 + lrow;
        dst[d * HDc + e] = acc[ct][r] + red[d * HDc + e];
      }
  }
}

// ============================================================================
// Kernel 2b: kv[bh] = sum over KSLICE partials (deterministic, no atomics)
// ============================================================================
__global__ __launch_bounds__(256) void kv_combine_kernel(
    const float* __restrict__ part, float* __restrict__ kv)
{
  const int bh = blockIdx.x;
#pragma unroll
  for (int j = 0; j < 16; ++j) {
    const int idx = threadIdx.x + j * 256;   // 0..4095
    float s = 0.f;
#pragma unroll
    for (int sl = 0; sl < KSLICE; ++sl)
      s += part[((long)sl * (Bc * NHc) + bh) * (HDc * HDc) + idx];
    kv[(long)bh * (HDc * HDc) + idx] = s;
  }
}

// ============================================================================
// Kernel 3: ksum[bh][d] = sum_t kf[b][t][h*64+d]  (4-way t split + LDS reduce)
// ============================================================================
__global__ __launch_bounds__(256) void ksum_kernel(const _Float16* __restrict__ kf,
                                                   float* __restrict__ ksum)
{
  __shared__ float red[256];
  const int bh = blockIdx.x;
  const int b = bh / NHc, h = bh % NHc;
  const int tid = threadIdx.x;
  const int d = tid & 63, tg = tid >> 6;
  const _Float16* p = kf + (long)b * Sc * Hc + (long)h * HDc + d;
  float s = 0.f;
  for (int t = tg * (Sc / 4); t < (tg + 1) * (Sc / 4); ++t)
    s += (float)p[(long)t * Hc];
  red[tid] = s;
  __syncthreads();
  if (tid < 64)
    ksum[bh * HDc + tid] = red[tid] + red[tid + 64] + red[tid + 128] + red[tid + 192];
}

// ============================================================================
// Kernel 4: attn[t][e] = (qf[t]·kv[:,e]) / (qf[t]·ksum + 1e-6)   per (b,h)
// ============================================================================
__global__ __launch_bounds__(256) void attn_apply_kernel(
    const _Float16* __restrict__ qf, const float* __restrict__ kv,
    const float* __restrict__ ksum, _Float16* __restrict__ attn)
{
  constexpr int LQ = 72;
  __shared__ __align__(16) _Float16 qs[128 * LQ];    // [row][d]
  __shared__ __align__(16) _Float16 kvT[HDc * LQ];   // [e][d]
  __shared__ float ks_s[HDc];
  __shared__ float norm_s[128];

  const int bh = blockIdx.y;
  const int b = bh / NHc, h = bh % NHc;
  const long headoff = (long)b * Sc * Hc + (long)h * HDc;
  const long t0 = (long)blockIdx.x * 128;

  const int tid = threadIdx.x, lane = tid & 31;
  const int wave = __builtin_amdgcn_readfirstlane(tid >> 5);
  const int kg = lane >> 4, lrow = lane & 15;

  // stage qf tile (128 x 64 f16): 4x 16B per thread, async when available
#pragma unroll
  for (int i = 0; i < 4; ++i) {
    const int id = tid + i * 256;
    const int row = id >> 3, c = (id & 7) * 8;
    copy16B_lds(qf + headoff + (t0 + row) * Hc + c, &qs[row * LQ + c]);
  }
  const float* kvp = kv + (long)bh * HDc * HDc;
#pragma unroll
  for (int j = 0; j < 16; ++j) {        // f32 kv -> f16, transposed to [e][d]
    const int idx = tid * 16 + j;
    const int d = idx >> 6, e = idx & 63;
    kvT[e * LQ + d] = (_Float16)kvp[idx];
  }
  if (tid < HDc) ks_s[tid] = ksum[bh * HDc + tid];
  async_join();
  __syncthreads();

  if (tid < 128) {                      // per-row normalization denominator
    float s = 0.f;
#pragma unroll 8
    for (int d = 0; d < HDc; ++d) s += (float)qs[tid * LQ + d] * ks_s[d];
    norm_s[tid] = 1.f / (s + 1e-6f);
  }
  __syncthreads();

  v8f acc[4] = {};
#pragma unroll
  for (int kst = 0; kst < 2; ++kst) {
    const v16h a = frag_a(&qs[(wave * 16 + lrow) * LQ + kst * 32], kg);
#pragma unroll
    for (int ct = 0; ct < 4; ++ct) {
      const v16h bf = frag_b(&kvT[(ct * 16 + lrow) * LQ + kst * 32], kg);
      acc[ct] = wmma_f16(a, bf, acc[ct]);
    }
  }
#pragma unroll
  for (int ct = 0; ct < 4; ++ct) {
#pragma unroll
    for (int r = 0; r < 8; ++r) {
      const int rloc = wave * 16 + kg * 8 + r;
      const float val = acc[ct][r] * norm_s[rloc];
      attn[headoff + (t0 + rloc) * Hc + ct * 16 + lrow] = (_Float16)val;
    }
  }
}

// ============================================================================
// Host launch.  Workspace layout:
//   qf|kf|v|attn  f16 [16384,1024] each          (4 x 32 MB)
//   W16 (q,k,v,o) f16 [1024,1024] each           (4 x 2 MB)
//   kv     f32 [64][64][64]                      (1 MB)
//   ksum   f32 [64][64]                          (16 KB)
//   kvpart f32 [16][64][64][64]                  (16 MB)      (~153 MB total)
// ============================================================================
extern "C" void kernel_launch(void* const* d_in, const int* in_sizes, int n_in,
                              void* d_out, int out_size, void* d_ws, size_t ws_size,
                              hipStream_t stream)
{
  const float* f_in = (const float*)d_in[0];
  const float* i_in = (const float*)d_in[1];
  const float* z_in = (const float*)d_in[2];
  // d_in[3] (o_in) is unused by the reference.
  const float* Wq = (const float*)d_in[4];
  const float* bq = (const float*)d_in[5];
  const float* Wk = (const float*)d_in[6];
  const float* bk = (const float*)d_in[7];
  const float* Wv = (const float*)d_in[8];
  const float* bv = (const float*)d_in[9];
  const float* Wo = (const float*)d_in[10];
  const float* bo = (const float*)d_in[11];
  float* out = (float*)d_out;

  const size_t NTH = (size_t)MTOK * Hc;
  const size_t HH  = (size_t)Hc * Hc;
  _Float16* qf   = (_Float16*)d_ws;
  _Float16* kf   = qf + NTH;
  _Float16* vvp  = kf + NTH;
  _Float16* attn = vvp + NTH;
  _Float16* w16q = attn + NTH;
  _Float16* w16k = w16q + HH;
  _Float16* w16v = w16k + HH;
  _Float16* w16o = w16v + HH;
  float* kvp    = (float*)(w16o + HH);
  float* ksum   = kvp + (size_t)Bc * NHc * HDc * HDc;
  float* kvpart = ksum + (size_t)Bc * NHc * HDc;

  const dim3 blk(256);
  const dim3 gw(HH / (256 * 4));          // 1024 blocks per weight cvt
  const dim3 gp(Hc / 128, MTOK / 128);    // (8, 128) blocks per GEMM

  hipLaunchKernelGGL(cvt_w_kernel, gw, blk, 0, stream, Wq, w16q);
  hipLaunchKernelGGL(cvt_w_kernel, gw, blk, 0, stream, Wk, w16k);
  hipLaunchKernelGGL(cvt_w_kernel, gw, blk, 0, stream, Wv, w16v);
  hipLaunchKernelGGL(cvt_w_kernel, gw, blk, 0, stream, Wo, w16o);

  hipLaunchKernelGGL((gemm_xwt<float, _Float16, true>),  gp, blk, 0, stream,
                     f_in, w16q, bq, qf, MTOK, Hc, Hc);
  hipLaunchKernelGGL((gemm_xwt<float, _Float16, true>),  gp, blk, 0, stream,
                     i_in, w16k, bk, kf, MTOK, Hc, Hc);
  hipLaunchKernelGGL((gemm_xwt<float, _Float16, false>), gp, blk, 0, stream,
                     z_in, w16v, bv, vvp, MTOK, Hc, Hc);

  hipLaunchKernelGGL(kv_reduce_kernel, dim3(Bc * NHc, KSLICE), blk, 0, stream,
                     kf, vvp, kvpart);
  hipLaunchKernelGGL(kv_combine_kernel, dim3(Bc * NHc), blk, 0, stream,
                     kvpart, kvp);
  hipLaunchKernelGGL(ksum_kernel, dim3(Bc * NHc), blk, 0, stream, kf, ksum);
  hipLaunchKernelGGL(attn_apply_kernel, dim3(Sc / 128, Bc * NHc), blk, 0, stream,
                     qf, kvp, ksum, attn);

  hipLaunchKernelGGL((gemm_xwt<_Float16, float, false>), gp, blk, 0, stream,
                     attn, w16o, bo, out, MTOK, Hc, Hc);
}
